// MultiHeadAttention_87067577025153
// MI455X (gfx1250) — matrix operand order repaired and checked by reading, not compiled, not run on hardware
//
#include <hip/hip_runtime.h>
#include <hip/hip_bf16.h>
#include <math.h>

#define B_  2
#define T_  2048
#define E_  1024
#define H_  16
#define D_  64

typedef __bf16 bf16;
typedef __attribute__((ext_vector_type(16))) __bf16 v16bf;
typedef __attribute__((ext_vector_type(8)))  __bf16 v8bf;
typedef __attribute__((ext_vector_type(8)))  float  v8f;

union F8  { v8f  v; float f[8]; };
union H16 { v16bf v; v8bf h[2]; };

// ---------------------------------------------------------------------------
// fragment loaders (CDNA5 WMMA 16x16x32 bf16 layouts, wave32)
// ---------------------------------------------------------------------------
__device__ __forceinline__ v16bf ld2x8(const bf16* p0, const bf16* p1) {
  H16 u;
  u.h[0] = *reinterpret_cast<const v8bf*>(p0);
  u.h[1] = *reinterpret_cast<const v8bf*>(p1);
  return u.v;
}

// A matrix 16x32 (MxK), row-major source with leading dim ld.
// lane l: row = mBase + l%16; chunks k = kBase + 8*(l>=16) and +16 more.
__device__ __forceinline__ v16bf ld_a(const bf16* base, int ld, int mBase,
                                      int kBase, int lane) {
  const bf16* p = base + (size_t)(mBase + (lane & 15)) * ld + kBase
                  + ((lane >> 4) << 3);
  return ld2x8(p, p + 16);
}

// B matrix 32x16 (KxN); source is "B transposed": rows indexed by N,
// contiguous over K with leading dim ld.  lane l: col = nBase + l%16,
// chunks k = kBase + 8*(l>=16) and +16 more.
__device__ __forceinline__ v16bf ld_b(const bf16* baseT, int ld, int nBase,
                                      int kBase, int lane) {
  const bf16* p = baseT + (size_t)(nBase + (lane & 15)) * ld + kBase
                  + ((lane >> 4) << 3);
  return ld2x8(p, p + 16);
}

__device__ __forceinline__ v8f wmma_bf16(v16bf a, v16bf b, v8f c) {
  return __builtin_amdgcn_wmma_f32_16x16x32_bf16(false, a, false, b,
                                                 (short)0, c, false, false);
}

__device__ __forceinline__ float rmax16(float x) {
  x = fmaxf(x, __shfl_xor(x, 1));
  x = fmaxf(x, __shfl_xor(x, 2));
  x = fmaxf(x, __shfl_xor(x, 4));
  x = fmaxf(x, __shfl_xor(x, 8));
  return x;  // xor masks < 16 never cross the 16-lane half-group
}
__device__ __forceinline__ float rsum16(float x) {
  x += __shfl_xor(x, 1);
  x += __shfl_xor(x, 2);
  x += __shfl_xor(x, 4);
  x += __shfl_xor(x, 8);
  return x;
}

// ---------------------------------------------------------------------------
// staging kernels
// ---------------------------------------------------------------------------
__global__ void k_f32_to_bf16(const float* __restrict__ in,
                              bf16* __restrict__ out, int n) {
  int i = blockIdx.x * blockDim.x + threadIdx.x;
  if (i < n) out[i] = (bf16)in[i];
}

// W[K][N] fp32 -> Wt[N][K] bf16
__global__ void k_transpose_bf16(const float* __restrict__ W,
                                 bf16* __restrict__ Wt, int K, int N) {
  int i = blockIdx.x * blockDim.x + threadIdx.x;
  if (i >= K * N) return;
  int k = i / N, n = i - k * N;
  Wt[(size_t)n * K + k] = (bf16)W[i];
}

// ---------------------------------------------------------------------------
// async global -> LDS staging of one 128x32 bf16 tile (8 KB).
// 512 x 16B chunks, 256 threads, 2 async b128 issues per thread.
// LDS byte offset = low 32 bits of the generic shared-memory pointer
// (aperture rule: generic LDS addr[31:0] == LDS offset).
// ---------------------------------------------------------------------------
__device__ __forceinline__ void async_stage_tile(const bf16* __restrict__ src,
                                                 int ldsrc, int rowBase, int k,
                                                 bf16 (*dst)[32], int tid) {
#pragma unroll
  for (int p = 0; p < 2; ++p) {
    int c   = p * 256 + tid;           // 16-byte chunk index, 0..511
    int row = c >> 2;
    int cp  = (c & 3) << 3;            // element offset within row (0,8,16,24)
    unsigned long long g =
        (unsigned long long)(uintptr_t)(src + (size_t)(rowBase + row) * ldsrc
                                        + k + cp);
    unsigned int l = (unsigned int)(uintptr_t)(&dst[row][cp]);
    asm volatile("global_load_async_to_lds_b128 %0, %1, off"
                 :: "v"(l), "v"(g) : "memory");
  }
}

// ---------------------------------------------------------------------------
// bf16 GEMM, M x 1024 x N.  block = 8 waves, block tile 128x128,
// wave tile 64x32 (4 M-frags x 2 N-frags).  A and B tiles double-buffered in
// LDS via async DMA; fragments fed from LDS.  mode 0: QKV epilogue, 1: proj.
// ---------------------------------------------------------------------------
__global__ __launch_bounds__(256)
void k_gemm(const bf16* __restrict__ A, const bf16* __restrict__ Bt,
            const float* __restrict__ bias, int mode,
            bf16* __restrict__ qb, bf16* __restrict__ kb,
            bf16* __restrict__ vtb, float* __restrict__ yout) {
  __shared__ __align__(16) bf16 ldsA[2][128][32];
  __shared__ __align__(16) bf16 ldsB[2][128][32];

  const int tid   = threadIdx.x;
  const int lane  = tid & 31;
  const int wave  = tid >> 5;
  const int mRow  = blockIdx.y * 128;            // block A tile base row
  const int nRow  = blockIdx.x * 128;            // block B tile base row (N)
  const int mLoc  = (wave >> 2) * 64;            // wave tile within block
  const int nLoc  = (wave & 3) * 32;

  v8f acc[4][2];
#pragma unroll
  for (int i = 0; i < 4; ++i)
#pragma unroll
    for (int j = 0; j < 2; ++j) acc[i][j] = (v8f){0.f,0.f,0.f,0.f,0.f,0.f,0.f,0.f};

  // prologue: stage k=0 tiles into buffer 0
  async_stage_tile(A,  E_, mRow, 0, ldsA[0], tid);
  async_stage_tile(Bt, E_, nRow, 0, ldsB[0], tid);
  asm volatile("s_wait_asynccnt 0" ::: "memory");
  __syncthreads();

  int cur = 0;
  for (int k = 0; k < E_; k += 32) {
    if (k + 32 < E_) {                           // prefetch next k-tile
      async_stage_tile(A,  E_, mRow, k + 32, ldsA[cur ^ 1], tid);
      async_stage_tile(Bt, E_, nRow, k + 32, ldsB[cur ^ 1], tid);
    }

    v16bf af[4], bfg[2];
#pragma unroll
    for (int mi = 0; mi < 4; ++mi) {
      const bf16* p = &ldsA[cur][mLoc + 16 * mi + (lane & 15)][(lane >> 4) << 3];
      af[mi] = ld2x8(p, p + 16);
    }
#pragma unroll
    for (int ni = 0; ni < 2; ++ni) {
      const bf16* p = &ldsB[cur][nLoc + 16 * ni + (lane & 15)][(lane >> 4) << 3];
      bfg[ni] = ld2x8(p, p + 16);
    }
#pragma unroll
    for (int mi = 0; mi < 4; ++mi)
#pragma unroll
      for (int ni = 0; ni < 2; ++ni)
        acc[mi][ni] = wmma_bf16(af[mi], bfg[ni], acc[mi][ni]);

    asm volatile("s_wait_asynccnt 0" ::: "memory");
    __syncthreads();
    cur ^= 1;
  }

  const int mBase = mRow + mLoc;
  const int nBase = nRow + nLoc;
#pragma unroll
  for (int mi = 0; mi < 4; ++mi)
#pragma unroll
    for (int ni = 0; ni < 2; ++ni) {
      F8 u; u.v = acc[mi][ni];
#pragma unroll
      for (int v = 0; v < 8; ++v) {
        int m = mBase + 16 * mi + ((lane >> 4) << 3) + v;
        int n = nBase + 16 * ni + (lane & 15);
        float val = u.f[v] + bias[n];
        if (mode == 0) {
          int sec = n >> 10, nn = n & 1023;
          int h = nn >> 6, d = nn & 63;
          int b = m >> 11, t = m & 2047;
          if (sec == 2) {
            vtb[(((size_t)b * H_ + h) * D_ + d) * T_ + t] = (bf16)val;
          } else {
            size_t idx = (((size_t)b * H_ + h) * T_ + t) * D_ + d;
            if (sec == 0) qb[idx] = (bf16)(val * 0.125f);  // fold 1/sqrt(D)
            else          kb[idx] = (bf16)val;
          }
        } else {
          yout[(size_t)m * E_ + n] = val;
        }
      }
    }
}

// ---------------------------------------------------------------------------
// flash attention: 1 wave = 16 queries, stream 32 keys/iter, online softmax.
// q [B,H,T,D] (prescaled), k [B,H,T,D], vt [B,H,D,T]; out aout bf16 [B,T,E].
// ---------------------------------------------------------------------------
__global__ __launch_bounds__(128)
void k_attn(const bf16* __restrict__ q, const bf16* __restrict__ kk,
            const bf16* __restrict__ vt, bf16* __restrict__ aout) {
  __shared__ __align__(16) bf16 ptile[4][16][32];

  const int lane = threadIdx.x & 31;
  const int wave = threadIdx.x >> 5;
  const int bh   = blockIdx.x;
  const int qTile = blockIdx.y * 64 + wave * 16;

  const bf16* qp = q  + (size_t)bh * T_ * D_;
  const bf16* kp = kk + (size_t)bh * T_ * D_;
  const bf16* vp = vt + (size_t)bh * D_ * T_;

  const v16bf qf0 = ld_a(qp, D_, qTile, 0,  lane);
  const v16bf qf1 = ld_a(qp, D_, qTile, 32, lane);

  v8f o[4];
#pragma unroll
  for (int i = 0; i < 4; ++i) o[i] = (v8f){0.f,0.f,0.f,0.f,0.f,0.f,0.f,0.f};
  float rowmax[8], rowsum[8];
#pragma unroll
  for (int v = 0; v < 8; ++v) { rowmax[v] = -__builtin_inff(); rowsum[v] = 0.f; }

  const v8f zf = (v8f){0.f,0.f,0.f,0.f,0.f,0.f,0.f,0.f};

  for (int kbp = 0; kbp < qTile + 16; kbp += 32) {
    // ---- S = Q K^T for 2 key sub-tiles of 16 -----------------------------
    F8 s0, s1;
    s0.v = wmma_bf16(qf0, ld_b(kp, D_, kbp,      0,  lane), zf);
    s0.v = wmma_bf16(qf1, ld_b(kp, D_, kbp,      32, lane), s0.v);
    s1.v = wmma_bf16(qf0, ld_b(kp, D_, kbp + 16, 0,  lane), zf);
    s1.v = wmma_bf16(qf1, ld_b(kp, D_, kbp + 16, 32, lane), s1.v);

    const bool needMask = (kbp + 31 > qTile);
    const int key0 = kbp + (lane & 15);
    float alpha[8];
#pragma unroll
    for (int v = 0; v < 8; ++v) {
      int qrow = qTile + ((lane >> 4) << 3) + v;
      if (needMask) {
        if (key0 > qrow)      s0.f[v] = -__builtin_inff();
        if (key0 + 16 > qrow) s1.f[v] = -__builtin_inff();
      }
      float tmax = rmax16(fmaxf(s0.f[v], s1.f[v]));
      float nm   = fmaxf(rowmax[v], tmax);
      alpha[v]   = __expf(rowmax[v] - nm);
      rowmax[v]  = nm;
      float p0 = __expf(s0.f[v] - nm);
      float p1 = __expf(s1.f[v] - nm);
      rowsum[v] = rowsum[v] * alpha[v] + rsum16(p0 + p1);
      int m = ((lane >> 4) << 3) + v;
      ptile[wave][m][lane & 15]        = (bf16)p0;
      ptile[wave][m][16 + (lane & 15)] = (bf16)p1;
    }
#pragma unroll
    for (int dti = 0; dti < 4; ++dti) {
      F8 u; u.v = o[dti];
#pragma unroll
      for (int v = 0; v < 8; ++v) u.f[v] *= alpha[v];
      o[dti] = u.v;
    }

    // intra-wave cross-lane LDS dependency: drain DS before re-reading
    asm volatile("s_wait_dscnt 0" ::: "memory");

    // P as A-fragment (16x32) from LDS
    const bf16* lrow = &ptile[wave][lane & 15][0];
    v16bf pf = ld2x8(lrow + ((lane >> 4) << 3), lrow + 16 + ((lane >> 4) << 3));

    // O += P V  (4 d-tiles of 16)
#pragma unroll
    for (int dti = 0; dti < 4; ++dti)
      o[dti] = wmma_bf16(pf, ld_b(vp, T_, dti * 16, kbp, lane), o[dti]);
  }

  // ---- normalize and scatter to [B,T,H*D] (bf16, feeds proj GEMM) --------
  const int b = bh >> 4, h = bh & 15;
#pragma unroll
  for (int dti = 0; dti < 4; ++dti) {
    F8 u; u.v = o[dti];
#pragma unroll
    for (int v = 0; v < 8; ++v) {
      int qrow = qTile + ((lane >> 4) << 3) + v;
      int d    = dti * 16 + (lane & 15);
      float val = u.f[v] / rowsum[v];
      aout[((size_t)(b * T_ + qrow)) * E_ + h * D_ + d] = (bf16)val;
    }
  }
}

// ---------------------------------------------------------------------------
extern "C" void kernel_launch(void* const* d_in, const int* in_sizes, int n_in,
                              void* d_out, int out_size, void* d_ws, size_t ws_size,
                              hipStream_t stream) {
  (void)in_sizes; (void)n_in; (void)out_size; (void)ws_size;
  const float* x      = (const float*)d_in[0];
  const float* W_attn = (const float*)d_in[1];
  const float* b_attn = (const float*)d_in[2];
  const float* W_proj = (const float*)d_in[3];
  const float* b_proj = (const float*)d_in[4];
  float* y = (float*)d_out;

  char* w = (char*)d_ws;
  const size_t nBTE = (size_t)B_ * T_ * E_;          // 4,194,304
  bf16* xb   = (bf16*)w;                 w += nBTE * 2;           // x bf16
  bf16* waT  = (bf16*)w;                 w += (size_t)3 * E_ * E_ * 2; // W_attn^T
  bf16* wpT  = (bf16*)w;                 w += (size_t)E_ * E_ * 2;     // W_proj^T
  bf16* qb   = (bf16*)w;                 w += nBTE * 2;
  bf16* kb   = (bf16*)w;                 w += nBTE * 2;
  bf16* vtb  = (bf16*)w;                 w += nBTE * 2;
  bf16* aoutb= (bf16*)w;                 w += nBTE * 2;

  // stage: convert / transpose to bf16
  k_f32_to_bf16<<<(int)((nBTE + 255) / 256), 256, 0, stream>>>(x, xb, (int)nBTE);
  {
    int n = 3 * E_ * E_;
    k_transpose_bf16<<<(n + 255) / 256, 256, 0, stream>>>(W_attn, waT, E_, 3 * E_);
  }
  {
    int n = E_ * E_;
    k_transpose_bf16<<<(n + 255) / 256, 256, 0, stream>>>(W_proj, wpT, E_, E_);
  }

  // QKV = x @ W_attn + b_attn, scattered to Q/K/V^T layouts
  k_gemm<<<dim3(3 * E_ / 128, (B_ * T_) / 128), 256, 0, stream>>>(
      xb, waT, b_attn, /*mode=*/0, qb, kb, vtb, nullptr);

  // flash attention
  k_attn<<<dim3(B_ * H_, T_ / 64), 128, 0, stream>>>(qb, kb, vtb, aoutb);

  // y = attn @ W_proj + b_proj
  k_gemm<<<dim3(E_ / 128, (B_ * T_) / 128), 256, 0, stream>>>(
      aoutb, wpT, b_proj, /*mode=*/1, nullptr, nullptr, nullptr, y);
}
